// SubCenterArcFaceLoss_90787018703217
// MI455X (gfx1250) — compile-verified, matrix operand order checked by reading, not run on hardware
//
#include <hip/hip_runtime.h>
#include <math.h>

// ---------------------------------------------------------------------------
// SubCenter-ArcFace on MI455X (gfx1250).
//   cosine = l2n(emb)[1024,512] @ l2n(weight)[150000,512]^T  -> max over K=3
//   -> margin on label column -> *S
// Strategy: fp32->bf16 normalize pre-pass (bf16 weights = 153.6MB, fits the
// 192MB L2, so only the first of 16 M-tile passes pays HBM), then a wave32
// WMMA GEMM using v_wmma_f32_16x16x32_bf16 with three accumulators per tile
// (one per subcenter), elementwise max, and a trig-free ArcFace margin
// epilogue:  cos(acos(x)+M) = x*cosM - sqrt(1-x^2)*sinM.
// Workspace use: (1024+150000)*512*2 B ~ 155 MB.
// ---------------------------------------------------------------------------

typedef __attribute__((ext_vector_type(16))) __bf16 v16bf;
typedef __attribute__((ext_vector_type(8)))  float  v8f;

#define B_SZ 1024
#define D_SZ 512
#define C_SZ 50000
#define K_SZ 3

__device__ __forceinline__ unsigned short f2bf(float f) {
  union { float f; unsigned u; } v; v.f = f;
  unsigned r = v.u + 0x7FFFu + ((v.u >> 16) & 1u);   // round-to-nearest-even
  return (unsigned short)(r >> 16);
}

// One block per row: L2-normalize a 512-wide fp32 row, emit bf16 bits.
__global__ __launch_bounds__(256)
void normalize_rows_bf16(const float* __restrict__ in,
                         unsigned short* __restrict__ out,
                         int nrows) {
  __shared__ float red[8];
  const int row = blockIdx.x;
  if (row >= nrows) return;
  const int t = threadIdx.x;                 // 256 threads, 2 elems each
  const float2 v = ((const float2*)(in + (size_t)row * D_SZ))[t];
  float ss = v.x * v.x + v.y * v.y;
  #pragma unroll
  for (int off = 16; off > 0; off >>= 1) ss += __shfl_down(ss, off, 32);
  const int lane = t & 31, wv = t >> 5;
  if (lane == 0) red[wv] = ss;
  __syncthreads();
  if (wv == 0) {
    float s = (lane < 8) ? red[lane] : 0.0f;
    #pragma unroll
    for (int off = 4; off > 0; off >>= 1) s += __shfl_down(s, off, 32);
    if (lane == 0) red[0] = s;
  }
  __syncthreads();
  const float inv = 1.0f / fmaxf(sqrtf(red[0]), 1e-12f);
  unsigned short* o = out + (size_t)row * D_SZ;
  o[2 * t]     = f2bf(v.x * inv);
  o[2 * t + 1] = f2bf(v.y * inv);
}

// Block: 8 waves. M-tile = 64 rows of B (A tile 64x512 bf16 = 64KB LDS,
// loaded once). Each wave computes 16 classes x 64 rows with 3 subcenter
// accumulators, 12 WMMAs per 32-deep D chunk, 16 chunks.
__global__ __launch_bounds__(256)
void arcface_wmma(const unsigned short* __restrict__ neb,   // [B,D] bf16 bits
                  const unsigned short* __restrict__ nwb,   // [C*K,D] bf16 bits
                  const int* __restrict__ labels,           // [B]
                  float* __restrict__ out) {                // [B,C]
  extern __shared__ unsigned char smem[];                   // 64*512*2 = 64KB
  unsigned short* lA = (unsigned short*)smem;

  const int mBase = blockIdx.y * 64;
  const int cBase = blockIdx.x * 128;
  const int t    = threadIdx.x;
  const int lane = t & 31;
  const int wv   = t >> 5;

  // Cooperative A-tile load: 64 rows x 512 bf16 = 4096 uint4.
  {
    const uint4* gA = (const uint4*)(neb + (size_t)mBase * D_SZ);
    uint4* sA = (uint4*)lA;
    #pragma unroll 4
    for (int i = t; i < 4096; i += 256) sA[i] = gA[i];
  }
  __syncthreads();

  v8f acc[4][3];
  #pragma unroll
  for (int m = 0; m < 4; ++m)
    #pragma unroll
    for (int k = 0; k < 3; ++k)
      acc[m][k] = (v8f){0.f, 0.f, 0.f, 0.f, 0.f, 0.f, 0.f, 0.f};

  const int myC   = cBase + wv * 16;
  const int col   = lane & 15;
  const int dHalf = (lane >> 4) * 16;       // lanes 16..31 take K=+16 half
  int cClamped = myC + col; if (cClamped >= C_SZ) cClamped = C_SZ - 1;
  size_t bRow[3];
  #pragma unroll
  for (int k = 0; k < 3; ++k)
    bRow[k] = ((size_t)cClamped * K_SZ + (size_t)k) * D_SZ;

  for (int ch = 0; ch < 16; ++ch) {
    const int dbase = ch * 32 + dHalf;
    v16bf a[4], b[3];
    #pragma unroll
    for (int m = 0; m < 4; ++m)
      a[m] = *(const v16bf*)(lA + ((m * 16 + col) * D_SZ + dbase));
    #pragma unroll
    for (int k = 0; k < 3; ++k) {
      b[k] = *(const v16bf*)(nwb + bRow[k] + dbase);
      if (ch < 15)   // warm GL2/L0 for the next chunk: global_prefetch_b8
        __builtin_prefetch((const void*)(nwb + bRow[k] + dbase + 32), 0, 1);
    }
    #pragma unroll
    for (int m = 0; m < 4; ++m)
      #pragma unroll
      for (int k = 0; k < 3; ++k)
        acc[m][k] = __builtin_amdgcn_wmma_f32_16x16x32_bf16(
            false, a[m], false, b[k], (short)0, acc[m][k], false, false);
  }

  // Epilogue: max over subcenters, ArcFace margin on the label column, store.
  // C/D layout: element (vgpr i, lane l): N = l&15, M = i + (l>=16 ? 8 : 0).
  // Margin without transcendentals: theta = acos(x) in [0,pi] =>
  //   cos(theta+M) = x*cosM - sqrt(1-x^2)*sinM.
  const float cosM = 0.87758256189f;   // cos(0.5)
  const float sinM = 0.47942553860f;   // sin(0.5)
  const int c = myC + col;
  if (c < C_SZ) {
    const int rHi = (lane >> 4) * 8;
    #pragma unroll
    for (int m = 0; m < 4; ++m) {
      #pragma unroll
      for (int i = 0; i < 8; ++i) {
        const float x = fmaxf(fmaxf(acc[m][0][i], acc[m][1][i]), acc[m][2][i]);
        const int brow = mBase + m * 16 + i + rHi;
        float val = x;
        if (labels[brow] == c) {
          const float cl = fminf(fmaxf(x, -1.0f + 1e-7f), 1.0f - 1e-7f);
          val = cl * cosM - sqrtf(fmaxf(1.0f - cl * cl, 0.0f)) * sinM;
        }
        out[(size_t)brow * C_SZ + c] = 50.0f * val;
      }
    }
  }
}

extern "C" void kernel_launch(void* const* d_in, const int* in_sizes, int n_in,
                              void* d_out, int out_size, void* d_ws, size_t ws_size,
                              hipStream_t stream) {
  const float* emb   = (const float*)d_in[0];   // [1024, 512] fp32
  const int* labels  = (const int*)d_in[1];     // [1024]
  const float* wgt   = (const float*)d_in[2];   // [50000, 3, 512] fp32
  float* out         = (float*)d_out;           // [1024, 50000] fp32

  unsigned short* neb = (unsigned short*)d_ws;                  // 1 MB
  unsigned short* nwb = neb + (size_t)B_SZ * D_SZ;              // 153.6 MB

  normalize_rows_bf16<<<B_SZ, 256, 0, stream>>>(emb, neb, B_SZ);
  normalize_rows_bf16<<<C_SZ * K_SZ, 256, 0, stream>>>(wgt, nwb, C_SZ * K_SZ);

  dim3 grid((C_SZ + 127) / 128, B_SZ / 64);     // 391 class-tiles x 16 M-tiles
  arcface_wmma<<<grid, 256, 64 * D_SZ * 2, stream>>>(neb, nwb, labels, out);
}
